// MultiHeadAttention_75763223101729
// MI455X (gfx1250) — matrix operand order repaired and checked by reading, not compiled, not run on hardware
//
#include <hip/hip_runtime.h>

// ---------------------------------------------------------------------------
// MultiHeadAttention forward for MI455X (gfx1250, wave32, WMMA 16x16x32 f16).
//
// Pipeline (all matmuls on v_wmma_f32_16x16x32_f16, f32 accumulate):
//   1) gemm_wmma x3 : Q/K/V projections, f32 in -> f16 (b,h,s,d) out.
//      1/sqrt(D_K) folded into the Q projection epilogue.
//   2) attn_wmma    : flash attention (online softmax), 128 queries/block,
//      64-key LDS tiles. K tile staged by the Tensor Data Mover (TDM,
//      tensor_load_to_lds) with hardware LDS padding reproducing the
//      conflict-free 144B pitch; V staged transposed in LDS.
//   3) gemm_wmma    : output projection, f16 in -> f32 d_out.
//
// Roofline: ~137 GFLOP total vs ~0.3 GB HBM traffic -> matrix-op bound;
// the 2048x2048 score matrix never touches HBM (flash softmax).
// Workspace: 4 x 8192x1024 f16 tensors = 64 MiB.
// ---------------------------------------------------------------------------

typedef __attribute__((ext_vector_type(16))) _Float16 v16h;
typedef __attribute__((ext_vector_type(8)))  float    v8f;
typedef __attribute__((ext_vector_type(4)))  unsigned int u32x4;
typedef __attribute__((ext_vector_type(8)))  int      i32x8;
typedef __attribute__((ext_vector_type(4)))  int      i32x4;

#define D_MODEL 1024
#define N_HEADS 16
#define D_K     64
#define BATCH   4
#define SEQ     2048
#define MROWS   (BATCH * SEQ)   // 8192

#ifndef __has_builtin
#define __has_builtin(x) 0
#endif
#if __has_builtin(__builtin_amdgcn_tensor_load_to_lds) && \
    __has_builtin(__builtin_amdgcn_s_wait_tensorcnt)
#define HAVE_TDM 1
#else
#define HAVE_TDM 0
#endif

union H16 { uint4 q[2]; _Float16 h[16]; };

__device__ __forceinline__ v16h ld_frag16(const unsigned char* p0,
                                          const unsigned char* p1) {
  union { uint4 q[2]; v16h v; } u;
  u.q[0] = *(const uint4*)p0;
  u.q[1] = *(const uint4*)p1;
  return u.v;
}

__device__ __forceinline__ v8f wmma_f16(v16h a, v16h b, v8f c) {
  return __builtin_amdgcn_wmma_f32_16x16x32_f16(false, a, false, b,
                                                (short)0, c, false, false);
}

#if HAVE_TDM
// 1-D TDM: copy nelem f16 from global to LDS, inserting 16B of LDS padding
// after every 128B (pad_interval code 4 = 16x8B, pad_amount code 3 = 4 DWORDs)
// -> reproduces the 144B conflict-free row pitch in hardware. D# per ISA
// CDNA5 ISA: count=1, type=2(image), data_size code 1 (2B).
__device__ __forceinline__ void tdm_load_f16_padded(unsigned lds_off,
                                                    const void* gptr,
                                                    int nelem) {
  const unsigned long long ga = (unsigned long long)gptr;
  u32x4 g0;
  g0.x = 1u;                                            // count=1 (valid)
  g0.y = lds_off;                                       // lds_addr
  g0.z = (unsigned)(ga & 0xFFFFFFFFu);                  // global_addr[31:0]
  g0.w = (unsigned)((ga >> 32) & 0x01FFFFFFu) | (2u << 30);  // [56:32], type=2
  i32x8 g1;
  g1[0] = (1 << 16) | (1 << 20) | (4 << 22) | (3 << 25);  // 2B elems, padding
  g1[1] = (nelem & 0xFFFF) << 16;       // tensor_dim0[15:0]
  g1[2] = (nelem >> 16) | (1 << 16);    // tensor_dim0[31:16], tensor_dim1=1
  g1[3] = (nelem & 0xFFFF) << 16;       // tile_dim0 (1-D tile)
  g1[4] = 0;                            // tile_dim1/2 unused
  g1[5] = nelem;                        // tensor_dim0_stride[31:0]
  g1[6] = 0;
  g1[7] = 0;
  const i32x4 z4 = {0, 0, 0, 0};
#if __clang_major__ >= 23
  const i32x8 z8 = {0, 0, 0, 0, 0, 0, 0, 0};
  __builtin_amdgcn_tensor_load_to_lds(g0, g1, z4, z4, z8, 0);
#else
  __builtin_amdgcn_tensor_load_to_lds(g0, g1, z4, z4, 0);
#endif
}
#endif

// ---------------------------------------------------------------------------
// C[8192,1024] = A[8192,1024] @ W^T + bias (W is [N,K] row-major, so both
// operands are k-contiguous). Block tile 128x128, K-step 32, 8 waves (2Mx4N),
// each wave 64x32 = 8 accumulators. LDS rows padded to 80B (conflict-free
// b128 fragment reads). f32->f16 conversion fused into LDS staging.
// ---------------------------------------------------------------------------
template <bool A_IS_F16, bool OUT_HEAD_F16>
__global__ __launch_bounds__(256)
void gemm_wmma_kernel(const void* __restrict__ Av,
                      const float* __restrict__ W,
                      const float* __restrict__ bias,
                      void* __restrict__ Cv, float outScale) {
  constexpr int K = D_MODEL, N = D_MODEL;
  constexpr int PITCH = 80;  // 32 f16 (64B) + 16B pad per LDS row
  __shared__ __align__(16) unsigned char As[128 * PITCH];
  __shared__ __align__(16) unsigned char Bs[128 * PITCH];

  const int tid  = threadIdx.x;
  const int lane = tid & 31, wave = tid >> 5;
  const int lh = lane & 15, hs = lane >> 4;
  const int wy = wave & 1;    // 2 waves over M (64 rows each)
  const int wx = wave >> 1;   // 4 waves over N (32 cols each)
  const int r    = tid >> 1;  // staging row 0..127
  const int half = tid & 1;   // 16-half segment of the 32-wide K slab

  const size_t am = (size_t)blockIdx.y * 128 + r;  // global A row
  const size_t bn = (size_t)blockIdx.x * 128 + r;  // global W row (= C col)

  v8f acc[4][2] = {};

  for (int kb = 0; kb < K; kb += 32) {
    __syncthreads();
    {  // stage A slab
      H16 u;
      if (A_IS_F16) {
        const uint4* p = (const uint4*)((const _Float16*)Av + am * K + kb + half * 16);
        u.q[0] = p[0]; u.q[1] = p[1];
      } else {
        const float4* p = (const float4*)((const float*)Av + am * K + kb + half * 16);
        const float4 f0 = p[0], f1 = p[1], f2 = p[2], f3 = p[3];
        const float fs[16] = {f0.x,f0.y,f0.z,f0.w, f1.x,f1.y,f1.z,f1.w,
                              f2.x,f2.y,f2.z,f2.w, f3.x,f3.y,f3.z,f3.w};
        #pragma unroll
        for (int i = 0; i < 16; ++i) u.h[i] = (_Float16)fs[i];
      }
      *(uint4*)(As + r * PITCH + half * 32)      = u.q[0];
      *(uint4*)(As + r * PITCH + half * 32 + 16) = u.q[1];
    }
    {  // stage B slab (rows of W, k-contiguous)
      const float4* p = (const float4*)(W + bn * K + kb + half * 16);
      const float4 f0 = p[0], f1 = p[1], f2 = p[2], f3 = p[3];
      const float fs[16] = {f0.x,f0.y,f0.z,f0.w, f1.x,f1.y,f1.z,f1.w,
                            f2.x,f2.y,f2.z,f2.w, f3.x,f3.y,f3.z,f3.w};
      H16 u;
      #pragma unroll
      for (int i = 0; i < 16; ++i) u.h[i] = (_Float16)fs[i];
      *(uint4*)(Bs + r * PITCH + half * 32)      = u.q[0];
      *(uint4*)(Bs + r * PITCH + half * 32 + 16) = u.q[1];
    }
    if (kb + 32 < K) {  // global_prefetch_b8 the next K slab
      if (A_IS_F16)
        __builtin_prefetch((const _Float16*)Av + am * K + kb + 32, 0, 0);
      else
        __builtin_prefetch((const float*)Av + am * K + kb + 32, 0, 0);
      __builtin_prefetch(W + bn * K + kb + 32, 0, 0);
    }
    __syncthreads();

    v16h a[4], b[2];
    #pragma unroll
    for (int i = 0; i < 4; ++i) {
      const int row = wy * 64 + i * 16 + lh;
      const unsigned char* base = As + row * PITCH + hs * 16;
      a[i] = ld_frag16(base, base + 32);
    }
    #pragma unroll
    for (int j = 0; j < 2; ++j) {
      const int col = wx * 32 + j * 16 + lh;
      const unsigned char* base = Bs + col * PITCH + hs * 16;
      b[j] = ld_frag16(base, base + 32);
    }
    #pragma unroll
    for (int i = 0; i < 4; ++i)
      #pragma unroll
      for (int j = 0; j < 2; ++j)
        acc[i][j] = wmma_f16(a[i], b[j], acc[i][j]);
  }

  // epilogue: bias, scale, scatter (C/D layout: VGPR g -> row g + 8*hs, col lh)
  #pragma unroll
  for (int i = 0; i < 4; ++i) {
    #pragma unroll
    for (int j = 0; j < 2; ++j) {
      const int gm0 = blockIdx.y * 128 + wy * 64 + i * 16 + hs * 8;
      const int gn  = blockIdx.x * 128 + wx * 32 + j * 16 + lh;
      const float bb = bias[gn];
      #pragma unroll
      for (int g = 0; g < 8; ++g) {
        const int gm = gm0 + g;
        const float v = (acc[i][j][g] + bb) * outScale;
        if (OUT_HEAD_F16) {
          const int b_ = gm >> 11, s_ = gm & (SEQ - 1);
          const int h_ = gn >> 6,  d_ = gn & (D_K - 1);
          ((_Float16*)Cv)[(((size_t)b_ * N_HEADS + h_) * SEQ + s_) * D_K + d_] =
              (_Float16)v;
        } else {
          ((float*)Cv)[(size_t)gm * N + gn] = v;
        }
      }
    }
  }
}

// ---------------------------------------------------------------------------
// Flash attention. Grid (SEQ/128, B*H). 8 waves x 16 query rows each.
// Q fragments loaded straight from global into WMMA A layout (Q pre-scaled).
// 64-key tiles: K staged row-major by TDM (hardware-padded 144B pitch),
// V transposed (d-major) in LDS. Online softmax with half-wave shuffles.
// ---------------------------------------------------------------------------
__global__ __launch_bounds__(256)
void attn_wmma_kernel(const _Float16* __restrict__ Qh,
                      const _Float16* __restrict__ Kh,
                      const _Float16* __restrict__ Vh,
                      _Float16* __restrict__ AO) {
  constexpr int PITCH = 144;  // 64 f16 (128B) + 16B pad
  __shared__ __align__(16) unsigned char Ks[64 * PITCH];      //  9 KiB
  __shared__ __align__(16) unsigned char Vt[64 * PITCH];      //  9 KiB
  __shared__ __align__(16) unsigned char Ps[8 * 16 * PITCH];  // 18 KiB

  const int tid = threadIdx.x;
  const int lane = tid & 31, wave = tid >> 5;
  const int lh = lane & 15, hs = lane >> 4;
  const int qtile = blockIdx.x, bh = blockIdx.y;
  const size_t seqbase = (size_t)bh * SEQ;

  // 16 query rows of this wave -> two A fragments (d = 0..31, 32..63)
  const int q0 = qtile * 128 + wave * 16;
  v16h qf[2];
  {
    const unsigned char* qrow =
        (const unsigned char*)(Qh + (seqbase + q0 + lh) * D_K);
    #pragma unroll
    for (int c = 0; c < 2; ++c) {
      const unsigned char* base = qrow + c * 64 + hs * 16;
      qf[c] = ld_frag16(base, base + 32);
    }
  }

  float m[8], l[8];
  #pragma unroll
  for (int g = 0; g < 8; ++g) { m[g] = -1e30f; l[g] = 0.0f; }
  v8f o[4] = {};

#if HAVE_TDM
  const unsigned ks_lds_off = (unsigned)(uintptr_t)(void*)&Ks[0];
#endif

  for (int kt = 0; kt < SEQ / 64; ++kt) {
    __syncthreads();
#if HAVE_TDM
    // --- stage K tile with the Tensor Data Mover (wave 0 issues the DMA) ---
    if (wave == 0) {
      tdm_load_f16_padded(ks_lds_off, Kh + (seqbase + kt * 64) * D_K, 64 * D_K);
      __builtin_amdgcn_s_wait_tensorcnt(0);
    }
#else
    {  // stage K tile: 64 rows x 64 halfs, d contiguous
      const int kr = tid >> 2, seg = tid & 3;
      const uint4* src =
          (const uint4*)(Kh + (seqbase + kt * 64 + kr) * D_K + seg * 16);
      uint4* dst = (uint4*)(Ks + kr * PITCH + seg * 32);
      dst[0] = src[0]; dst[1] = src[1];
    }
#endif
    {  // stage V^T tile: d-major rows, k contiguous
      const int kr = tid >> 2, dseg = tid & 3;
      H16 u;
      const uint4* src =
          (const uint4*)(Vh + (seqbase + kt * 64 + kr) * D_K + dseg * 16);
      u.q[0] = src[0]; u.q[1] = src[1];
      #pragma unroll
      for (int t = 0; t < 16; ++t)
        *(_Float16*)(Vt + (dseg * 16 + t) * PITCH + kr * 2) = u.h[t];
    }
    if (kt + 1 < SEQ / 64) {  // prefetch next tiles (one row per thread)
      __builtin_prefetch(Kh + (seqbase + (kt + 1) * 64 + (tid >> 2)) * D_K, 0, 0);
      __builtin_prefetch(Vh + (seqbase + (kt + 1) * 64 + (tid >> 2)) * D_K, 0, 0);
    }
    __syncthreads();

    // scores S = (Q/sqrt(dk)) @ K^T : 16 x 64 per wave
    v8f s[4];
    #pragma unroll
    for (int j = 0; j < 4; ++j) {
      v8f t = {};
      #pragma unroll
      for (int c = 0; c < 2; ++c) {
        const unsigned char* base = Ks + (j * 16 + lh) * PITCH + c * 64 + hs * 16;
        t = wmma_f16(qf[c], ld_frag16(base, base + 32), t);
      }
      s[j] = t;
    }

    // online softmax: row g+8*hs lives across one 16-lane half
    float fac[8];
    #pragma unroll
    for (int g = 0; g < 8; ++g) {
      float v = s[0][g];
      #pragma unroll
      for (int j = 1; j < 4; ++j) v = fmaxf(v, s[j][g]);
      #pragma unroll
      for (int sh = 1; sh < 16; sh <<= 1) v = fmaxf(v, __shfl_xor(v, sh, 32));
      const float nm = fmaxf(m[g], v);
      fac[g] = __expf(m[g] - nm);
      m[g] = nm;
    }
    #pragma unroll
    for (int g = 0; g < 8; ++g) {
      float rs = 0.0f;
      #pragma unroll
      for (int j = 0; j < 4; ++j) {
        const float p = __expf(s[j][g] - m[g]);
        s[j][g] = p;
        rs += p;
      }
      #pragma unroll
      for (int sh = 1; sh < 16; sh <<= 1) rs += __shfl_xor(rs, sh, 32);
      l[g] = l[g] * fac[g] + rs;
      #pragma unroll
      for (int dt = 0; dt < 4; ++dt) o[dt][g] *= fac[g];
    }

    // stage P (f16) into wave-private LDS slab for the P@V A-fragments
    unsigned char* pw = Ps + wave * (16 * PITCH);
    #pragma unroll
    for (int j = 0; j < 4; ++j)
      #pragma unroll
      for (int g = 0; g < 8; ++g)
        *(_Float16*)(pw + (g + hs * 8) * PITCH + (j * 16 + lh) * 2) =
            (_Float16)s[j][g];
    // same-wave DS ops retire in order; fence stops compiler/issue reordering
    asm volatile("s_wait_dscnt 0" ::: "memory");

    // O += P @ V
    #pragma unroll
    for (int c = 0; c < 2; ++c) {
      const unsigned char* pa = pw + lh * PITCH + c * 64 + hs * 16;
      const v16h pf = ld_frag16(pa, pa + 32);
      #pragma unroll
      for (int dt = 0; dt < 4; ++dt) {
        const unsigned char* vb = Vt + (dt * 16 + lh) * PITCH + c * 64 + hs * 16;
        o[dt] = wmma_f16(pf, ld_frag16(vb, vb + 32), o[dt]);
      }
    }
  }

  // normalize, scatter to (b, s, h*64+d) f16 for the output projection GEMM
  const int b_ = bh >> 4, h_ = bh & 15;
  #pragma unroll
  for (int g = 0; g < 8; ++g) {
    const float inv = 1.0f / l[g];
    const int srow = q0 + g + hs * 8;
    _Float16* dst = AO + ((size_t)b_ * SEQ + srow) * D_MODEL + h_ * D_K;
    #pragma unroll
    for (int dt = 0; dt < 4; ++dt)
      dst[dt * 16 + lh] = (_Float16)(o[dt][g] * inv);
  }
}

extern "C" void kernel_launch(void* const* d_in, const int* in_sizes, int n_in,
                              void* d_out, int out_size, void* d_ws, size_t ws_size,
                              hipStream_t stream) {
  (void)in_sizes; (void)n_in; (void)out_size; (void)ws_size;
  const float* query = (const float*)d_in[0];
  const float* key_  = (const float*)d_in[1];
  const float* value = (const float*)d_in[2];
  const float* Wq = (const float*)d_in[3];
  const float* bq = (const float*)d_in[4];
  const float* Wk = (const float*)d_in[5];
  const float* bk = (const float*)d_in[6];
  const float* Wv = (const float*)d_in[7];
  const float* bv = (const float*)d_in[8];
  const float* Wo = (const float*)d_in[9];
  const float* bo = (const float*)d_in[10];

  const size_t ntok = (size_t)MROWS * D_MODEL;  // 8192*1024 elements
  _Float16* Qh = (_Float16*)d_ws;               // 16 MiB each, 64 MiB total
  _Float16* Kh = Qh + ntok;
  _Float16* Vh = Kh + ntok;
  _Float16* AO = Vh + ntok;

  const dim3 blk(256);
  const dim3 ggrid(D_MODEL / 128, MROWS / 128);  // (8, 64)
  const float qscale = 0.125f;                   // 1/sqrt(D_K), folded into Q

  gemm_wmma_kernel<false, true><<<ggrid, blk, 0, stream>>>(query, Wq, bq, Qh, qscale);
  gemm_wmma_kernel<false, true><<<ggrid, blk, 0, stream>>>(key_,  Wk, bk, Kh, 1.0f);
  gemm_wmma_kernel<false, true><<<ggrid, blk, 0, stream>>>(value, Wv, bv, Vh, 1.0f);

  attn_wmma_kernel<<<dim3(SEQ / 128, BATCH * N_HEADS), blk, 0, stream>>>(Qh, Kh, Vh, AO);

  gemm_wmma_kernel<true, false><<<ggrid, blk, 0, stream>>>(AO, Wo, bo, d_out, 1.0f);
}
